// Double_DoubleAttentionLayer_norm_64321430225068
// MI455X (gfx1250) — matrix operand order repaired and checked by reading, not compiled, not run on hardware
//
#include <hip/hip_runtime.h>
#include <hip/hip_bf16.h>
#include <stdint.h>

// ---------------------------------------------------------------------------
// MI455X (gfx1250) implementation of the double-attention block.
// All heavy GEMMs use v_wmma_f32_16x16x32_bf16 (wave32, 16x16 tiles).
// Branch-2 uses the algebraic factorization Z_S = (softV_S softB_S^T) A_S,
// avoiding the [b,3136,3136] intermediate entirely.
// Register-blocked: conv3x3 does 64Mx32N per wave (1.5 b128 loads / WMMA),
// QKV GEMM does 32Mx32N per wave (2 b128 loads / WMMA).
// ---------------------------------------------------------------------------

typedef __bf16 bf16_t;
typedef __attribute__((ext_vector_type(16))) __bf16 v16bf;
typedef __attribute__((ext_vector_type(8)))  float  v8f;

#define HWP   3136
#define HD    56
#define CDIM  512
#define CM    128
#define CN    64
#define BATCH 8
#define MQ    640           // 128+64+64+128+128+128 fused 1x1-conv rows
#define KTAP  4608          // 9 taps * 512 ic (im2col K for 3x3 conv)

// ---- WMMA fragment loaders (contiguous K order, wave32) -------------------
// A (16x32 bf16, [M][K] row-major): lane m=L&15, half=L>>4;
//   elems 0..15 -> K = half*16 + 0..15
static __device__ __forceinline__ v16bf frag_a(const bf16_t* tile, int ldk,
                                               int k0, int lane) {
  int m = lane & 15, half = lane >> 4;
  const bf16_t* p = tile + (size_t)m * ldk + k0 + half * 16;
  union { v16bf v; uint4 q[2]; } u;
  u.q[0] = *(const uint4*)(p);
  u.q[1] = *(const uint4*)(p + 8);
  return u.v;
}
// B (32x16 bf16, stored [N][K]): lane n=L&15, kh=L>>4; K = kh*16 + 0..15
static __device__ __forceinline__ v16bf frag_b(const bf16_t* tile, int ldk,
                                               int k0, int lane) {
  int n = lane & 15, kh = lane >> 4;
  const bf16_t* p = tile + (size_t)n * ldk + k0 + kh * 16;
  union { v16bf v; uint4 q[2]; } u;
  u.q[0] = *(const uint4*)(p);
  u.q[1] = *(const uint4*)(p + 8);
  return u.v;
}
static __device__ __forceinline__ v8f wmma_bf16(v16bf a, v16bf b, v8f c) {
  return __builtin_amdgcn_wmma_f32_16x16x32_bf16(false, a, false, b,
                                                 (short)0, c, false, false);
}
static __device__ __forceinline__ uint4 pack8(v8f a) {
  union { uint4 q; bf16_t h[8]; } u;
#pragma unroll
  for (int r = 0; r < 8; ++r) u.h[r] = (bf16_t)a[r];
  return u.q;
}
static __device__ __forceinline__ v8f vzero() {
  v8f z = {0.f,0.f,0.f,0.f,0.f,0.f,0.f,0.f};
  return z;
}

// ---------------------------------------------------------------------------
// Prep kernels: pack weights to bf16 in GEMM-friendly layouts.
// ---------------------------------------------------------------------------
__global__ void k_pack_wcat(const float* WA, const float* WB, const float* WV,
                            const float* WAs, const float* WBs, const float* WVs,
                            const float* bA, const float* bB, const float* bV,
                            const float* bAs, const float* bBs, const float* bVs,
                            bf16_t* Wcat, float* bcat) {
  int idx = blockIdx.x * 256 + threadIdx.x;
  if (idx < MQ * CDIM) {
    int m = idx / CDIM, k = idx % CDIM;
    float v;
    if      (m < 128) v = WA [(m      ) * CDIM + k];
    else if (m < 192) v = WB [(m - 128) * CDIM + k];
    else if (m < 256) v = WV [(m - 192) * CDIM + k];
    else if (m < 384) v = WAs[(m - 256) * CDIM + k];
    else if (m < 512) v = WBs[(m - 384) * CDIM + k];
    else              v = WVs[(m - 512) * CDIM + k];
    Wcat[idx] = (bf16_t)v;
  }
  if (idx < MQ) {
    int m = idx; float v;
    if      (m < 128) v = bA [m      ];
    else if (m < 192) v = bB [m - 128];
    else if (m < 256) v = bV [m - 192];
    else if (m < 384) v = bAs[m - 256];
    else if (m < 512) v = bBs[m - 384];
    else              v = bVs[m - 512];
    bcat[m] = v;
  }
}

__global__ void k_pack_we(const float* WE, bf16_t* WEb) {
  int idx = blockIdx.x * 256 + threadIdx.x;
  if (idx < CDIM * CM) WEb[idx] = (bf16_t)WE[idx];
}

// Wout [oc][ic][3][3] -> Woutb [oc][tap][ic]  (K = tap*512 + ic)
__global__ void k_pack_wout(const float* Wout, bf16_t* Woutb) {
  int idx = blockIdx.x * 256 + threadIdx.x;
  if (idx < CDIM * KTAP) {
    int oc = idx / KTAP, r = idx % KTAP;
    int tap = r / CDIM, ic = r % CDIM;
    Woutb[idx] = (bf16_t)Wout[((size_t)(oc * CDIM + ic)) * 9 + tap];
  }
}

__global__ void k_bnprep(const float* g, const float* beta, const float* mean,
                         const float* var, float* scale, float* shift) {
  int idx = blockIdx.x * 256 + threadIdx.x;
  if (idx < CDIM) {
    float s = g[idx] * rsqrtf(var[idx] + 1e-5f);
    scale[idx] = s;
    shift[idx] = beta[idx] - mean[idx] * s;
  }
}

// x [b][c][hw] f32 -> xT [b][hw][c] bf16 (32x32 LDS tiles)
__global__ __launch_bounds__(256) void k_transpose_x(const float* x, bf16_t* xT) {
  __shared__ float t[32][33];
  int tx = threadIdx.x, ty = threadIdx.y;
  int pix0 = blockIdx.x * 32, c0 = blockIdx.y * 32, b = blockIdx.z;
#pragma unroll
  for (int i = 0; i < 4; ++i) {
    int c = c0 + ty + i * 8;
    t[ty + i * 8][tx] = x[((size_t)(b * CDIM + c)) * HWP + pix0 + tx];
  }
  __syncthreads();
#pragma unroll
  for (int i = 0; i < 4; ++i) {
    int pl = ty + i * 8;
    xT[((size_t)(b * HWP + pix0 + pl)) * CDIM + c0 + tx] = (bf16_t)t[tx][pl];
  }
}

// ---------------------------------------------------------------------------
// Fused 1x1 convs: Y[640 x 25088] = Wcat * xT^T + bcat  (K=512)
// 2Mx2N register blocking: 4 WMMAs per k-step, 8 b128 operand loads.
// Outputs routed to six buffers; A_S/B_S/V_S stored transposed [pix][m].
// ---------------------------------------------------------------------------
static __device__ __forceinline__ void qkv_store(
    int m0, int pix, int b, v8f acc, const float* bcat,
    bf16_t* YA, bf16_t* YB, bf16_t* YV,
    bf16_t* AsT, bf16_t* BsT, bf16_t* VsT) {
#pragma unroll
  for (int r = 0; r < 8; ++r) acc[r] += bcat[m0 + r];
  if (m0 < 128) {
#pragma unroll
    for (int r = 0; r < 8; ++r)
      YA[((size_t)(b * CM + m0 + r)) * HWP + pix] = (bf16_t)acc[r];
  } else if (m0 < 192) {
#pragma unroll
    for (int r = 0; r < 8; ++r)
      YB[((size_t)(b * CN + (m0 - 128) + r)) * HWP + pix] = (bf16_t)acc[r];
  } else if (m0 < 256) {
#pragma unroll
    for (int r = 0; r < 8; ++r)
      YV[((size_t)(b * CN + (m0 - 192) + r)) * HWP + pix] = (bf16_t)acc[r];
  } else if (m0 < 384) {
    *(uint4*)(AsT + ((size_t)(b * HWP + pix)) * CM + (m0 - 256)) = pack8(acc);
  } else if (m0 < 512) {
    *(uint4*)(BsT + ((size_t)(b * HWP + pix)) * CM + (m0 - 384)) = pack8(acc);
  } else {
    *(uint4*)(VsT + ((size_t)(b * HWP + pix)) * CM + (m0 - 512)) = pack8(acc);
  }
}

__global__ __launch_bounds__(128) void k_gemm_qkv(
    const bf16_t* xT, const bf16_t* Wcat, const float* bcat,
    bf16_t* YA, bf16_t* YB, bf16_t* YV,
    bf16_t* AsT, bf16_t* BsT, bf16_t* VsT) {
  int tid = threadIdx.x, w = tid >> 5, lane = tid & 31;
  int n0 = blockIdx.x * 32, b = blockIdx.z;
  int mslot = blockIdx.y * 4 + w;          // 0..19, 32 M-rows each
  const bf16_t* A0 = Wcat + (size_t)mslot * 32 * CDIM;
  const bf16_t* A1 = A0 + (size_t)16 * CDIM;
  const bf16_t* B0 = xT + ((size_t)b * HWP + n0) * CDIM;
  const bf16_t* B1 = B0 + (size_t)16 * CDIM;
  v8f acc00 = vzero(), acc01 = vzero(), acc10 = vzero(), acc11 = vzero();
  for (int kk = 0; kk < CDIM; kk += 32) {
    v16bf a0 = frag_a(A0, CDIM, kk, lane);
    v16bf a1 = frag_a(A1, CDIM, kk, lane);
    v16bf b0 = frag_b(B0, CDIM, kk, lane);
    v16bf b1 = frag_b(B1, CDIM, kk, lane);
    acc00 = wmma_bf16(a0, b0, acc00);
    acc01 = wmma_bf16(a0, b1, acc01);
    acc10 = wmma_bf16(a1, b0, acc10);
    acc11 = wmma_bf16(a1, b1, acc11);
  }
  int ncol = lane & 15, half = lane >> 4;
  int mA = mslot * 32 + half * 8, mB = mA + 16;
  int p0 = n0 + ncol, p1 = p0 + 16;
  qkv_store(mA, p0, b, acc00, bcat, YA, YB, YV, AsT, BsT, VsT);
  qkv_store(mA, p1, b, acc01, bcat, YA, YB, YV, AsT, BsT, VsT);
  qkv_store(mB, p0, b, acc10, bcat, YA, YB, YV, AsT, BsT, VsT);
  qkv_store(mB, p1, b, acc11, bcat, YA, YB, YV, AsT, BsT, VsT);
}

// ---------------------------------------------------------------------------
// Softmaxes
// ---------------------------------------------------------------------------
__global__ __launch_bounds__(256) void k_softmax_row(const bf16_t* in, bf16_t* out) {
  __shared__ float red[256];
  int row = blockIdx.x, tid = threadIdx.x;
  const bf16_t* p = in + (size_t)row * HWP;
  float mx = -1e30f;
  for (int i = tid; i < HWP; i += 256) mx = fmaxf(mx, (float)p[i]);
  red[tid] = mx; __syncthreads();
  for (int s = 128; s > 0; s >>= 1) {
    if (tid < s) red[tid] = fmaxf(red[tid], red[tid + s]);
    __syncthreads();
  }
  float m = red[0]; __syncthreads();
  float sum = 0.f;
  for (int i = tid; i < HWP; i += 256) sum += __expf((float)p[i] - m);
  red[tid] = sum; __syncthreads();
  for (int s = 128; s > 0; s >>= 1) {
    if (tid < s) red[tid] += red[tid + s];
    __syncthreads();
  }
  float inv = 1.0f / red[0];
  bf16_t* o = out + (size_t)row * HWP;
  for (int i = tid; i < HWP; i += 256) o[i] = (bf16_t)(__expf((float)p[i] - m) * inv);
}

// softV: softmax over c_n=64 per pixel; in [b][64][hw], out transposed [b][hw][64].
__global__ void k_softmax_cn64(const bf16_t* YV, bf16_t* softVt) {
  int gid = blockIdx.x * 256 + threadIdx.x;
  if (gid >= BATCH * HWP) return;
  int b = gid / HWP, pix = gid % HWP;
  const bf16_t* p = YV + (size_t)b * CN * HWP + pix;
  float mx = -1e30f;
  for (int c = 0; c < CN; ++c) mx = fmaxf(mx, (float)p[(size_t)c * HWP]);
  float sum = 0.f;
  for (int c = 0; c < CN; ++c) sum += __expf((float)p[(size_t)c * HWP] - mx);
  float inv = 1.0f / sum;
  bf16_t* o = softVt + ((size_t)b * HWP + pix) * CN;
  for (int c = 0; c < CN; ++c)
    o[c] = (bf16_t)(__expf((float)p[(size_t)c * HWP] - mx) * inv);
}

// softmax over 128 channels per pixel; in [b][hw][128], out [b][128][hw].
__global__ void k_softmax128(const bf16_t* inT, bf16_t* outM) {
  int gid = blockIdx.x * 256 + threadIdx.x;
  if (gid >= BATCH * HWP) return;
  int b = gid / HWP, pix = gid % HWP;
  const bf16_t* p = inT + ((size_t)b * HWP + pix) * CM;
  float mx = -1e30f;
  for (int c = 0; c < CM; ++c) mx = fmaxf(mx, (float)p[c]);
  float sum = 0.f;
  for (int c = 0; c < CM; ++c) sum += __expf((float)p[c] - mx);
  float inv = 1.0f / sum;
  for (int c = 0; c < CM; ++c)
    outM[((size_t)(b * CM + c)) * HWP + pix] = (bf16_t)(__expf((float)p[c] - mx) * inv);
}

// ---------------------------------------------------------------------------
// Attention matmuls
// ---------------------------------------------------------------------------
// G[b,128,64] = A[b,128,hw] x softB[b,64,hw]^T   (K = hw)
__global__ __launch_bounds__(256) void k_gemm_G(const bf16_t* YA,
                                                const bf16_t* softB, bf16_t* G) {
  int tid = threadIdx.x, w = tid >> 5, lane = tid & 31;
  int ntile = blockIdx.x, b = blockIdx.z, mtile = w;
  const bf16_t* A = YA + ((size_t)b * CM + mtile * 16) * HWP;
  const bf16_t* B = softB + ((size_t)b * CN + ntile * 16) * HWP;
  v8f acc = vzero();
  for (int kk = 0; kk < HWP; kk += 32)
    acc = wmma_bf16(frag_a(A, HWP, kk, lane), frag_b(B, HWP, kk, lane), acc);
  int ncol = lane & 15, half = lane >> 4;
#pragma unroll
  for (int r = 0; r < 8; ++r)
    G[((size_t)(b * CM + mtile * 16 + half * 8 + r)) * CN + ntile * 16 + ncol] =
        (bf16_t)acc[r];
}

// MS[b,128,128] = softV_S[b,128,hw] x softB_S[b,128,hw]^T   (Gram factorization)
__global__ __launch_bounds__(256) void k_gemm_MS(const bf16_t* VsM,
                                                 const bf16_t* BsM, bf16_t* MS) {
  int tid = threadIdx.x, w = tid >> 5, lane = tid & 31;
  int ntile = blockIdx.x, b = blockIdx.z, mtile = w;
  const bf16_t* A = VsM + ((size_t)b * CM + mtile * 16) * HWP;
  const bf16_t* B = BsM + ((size_t)b * CM + ntile * 16) * HWP;
  v8f acc = vzero();
  for (int kk = 0; kk < HWP; kk += 32)
    acc = wmma_bf16(frag_a(A, HWP, kk, lane), frag_b(B, HWP, kk, lane), acc);
  int ncol = lane & 15, half = lane >> 4;
#pragma unroll
  for (int r = 0; r < 8; ++r)
    MS[((size_t)(b * CM + mtile * 16 + half * 8 + r)) * CM + ntile * 16 + ncol] =
        (bf16_t)acc[r];
}

// Zt[b,hw,128]^T = G[b,128,64] x softVt[b,hw,64]^T   (K = 64)
__global__ __launch_bounds__(256) void k_gemm_Z(const bf16_t* G,
                                                const bf16_t* softVt, bf16_t* Zt) {
  int tid = threadIdx.x, w = tid >> 5, lane = tid & 31;
  int ntile = blockIdx.x, b = blockIdx.z, mtile = w;
  const bf16_t* A = G + (size_t)b * CM * CN + mtile * 16 * CN;
  const bf16_t* B = softVt + ((size_t)b * HWP + ntile * 16) * CN;
  v8f acc = vzero();
  for (int kk = 0; kk < CN; kk += 32)
    acc = wmma_bf16(frag_a(A, CN, kk, lane), frag_b(B, CN, kk, lane), acc);
  int ncol = lane & 15, half = lane >> 4;
  int pix = ntile * 16 + ncol;
  *(uint4*)(Zt + ((size_t)(b * HWP + pix)) * CM + mtile * 16 + half * 8) = pack8(acc);
}

// Zst[b,hw,128]^T = MS[b,128,128] x AsT[b,hw,128]^T   (K = 128)
__global__ __launch_bounds__(256) void k_gemm_ZS(const bf16_t* MS,
                                                 const bf16_t* AsT, bf16_t* Zst) {
  int tid = threadIdx.x, w = tid >> 5, lane = tid & 31;
  int ntile = blockIdx.x, b = blockIdx.z, mtile = w;
  const bf16_t* A = MS + (size_t)b * CM * CM + mtile * 16 * CM;
  const bf16_t* B = AsT + ((size_t)b * HWP + ntile * 16) * CM;
  v8f acc = vzero();
  for (int kk = 0; kk < CM; kk += 32)
    acc = wmma_bf16(frag_a(A, CM, kk, lane), frag_b(B, CM, kk, lane), acc);
  int ncol = lane & 15, half = lane >> 4;
  int pix = ntile * 16 + ncol;
  *(uint4*)(Zst + ((size_t)(b * HWP + pix)) * CM + mtile * 16 + half * 8) = pack8(acc);
}

// ---------------------------------------------------------------------------
// Fused E-stage: Esum[pix][oc] = relu(bn(WE*Z+bE)) + relu(bn(WE*Z_S+bE)) + x
// B-panels (16 pix x 128 ch, 4KB each) staged in LDS via async load-to-LDS.
// ---------------------------------------------------------------------------
__global__ __launch_bounds__(256) void k_gemm_E(
    const bf16_t* Zt, const bf16_t* Zst, const bf16_t* WEb, const float* bE,
    const float* scale, const float* shift, const bf16_t* xT, bf16_t* Esum) {
  __shared__ bf16_t sZ [16 * CM];
  __shared__ bf16_t sZs[16 * CM];
  int tid = threadIdx.x, w = tid >> 5, lane = tid & 31;
  int ntile = blockIdx.x, b = blockIdx.z;
  int mtile = blockIdx.y * 8 + w;
  const bf16_t* B1 = Zt  + ((size_t)b * HWP + ntile * 16) * CM;
  const bf16_t* B2 = Zst + ((size_t)b * HWP + ntile * 16) * CM;
  {
    // Each of the 256 threads async-copies one 16B chunk per panel (4KB/panel).
    uint32_t l1 = (uint32_t)(uintptr_t)(&sZ[0])  + tid * 16;
    uint64_t g1 = (uint64_t)(uintptr_t)(B1) + tid * 16;
    asm volatile("global_load_async_to_lds_b128 %0, %1, off"
                 :: "v"(l1), "v"(g1) : "memory");
    uint32_t l2 = (uint32_t)(uintptr_t)(&sZs[0]) + tid * 16;
    uint64_t g2 = (uint64_t)(uintptr_t)(B2) + tid * 16;
    asm volatile("global_load_async_to_lds_b128 %0, %1, off"
                 :: "v"(l2), "v"(g2) : "memory");
    asm volatile("s_wait_asynccnt 0" ::: "memory");
  }
  __syncthreads();
  const bf16_t* A = WEb + (size_t)mtile * 16 * CM;
  v8f acc1 = vzero();
  v8f acc2 = vzero();
  for (int kk = 0; kk < CM; kk += 32) {
    v16bf a = frag_a(A, CM, kk, lane);
    acc1 = wmma_bf16(a, frag_b(sZ,  CM, kk, lane), acc1);
    acc2 = wmma_bf16(a, frag_b(sZs, CM, kk, lane), acc2);
  }
  int ncol = lane & 15, half = lane >> 4;
  int pix = ntile * 16 + ncol;
  int m0  = mtile * 16 + half * 8;
  union { uint4 q; bf16_t h[8]; } xu;
  xu.q = *(const uint4*)(xT + ((size_t)(b * HWP + pix)) * CDIM + m0);
  v8f outv;
#pragma unroll
  for (int r = 0; r < 8; ++r) {
    int oc = m0 + r;
    float t1 = (acc1[r] + bE[oc]) * scale[oc] + shift[oc];
    float t2 = (acc2[r] + bE[oc]) * scale[oc] + shift[oc];
    outv[r] = fmaxf(t1, 0.f) + fmaxf(t2, 0.f) + (float)xu.h[r];
  }
  *(uint4*)(Esum + ((size_t)(b * HWP + pix)) * CDIM + m0) = pack8(outv);
}

// ---------------------------------------------------------------------------
// 3x3 conv as implicit GEMM: out[oc][pix] = sum_{tap,ic} Woutb * Esum + bout
// 118 GFLOP — the dominant kernel. Register-blocked 64Mx32N per wave:
// 8 WMMAs per k-step from 4 A-frags + 2 B-frags (12 b128 loads).
// Block of 8 waves covers all M=512 for one 32-pixel strip, so all waves
// share identical B-fragments (WGP$ reuse). K = 9 taps x 512 ic.
// ---------------------------------------------------------------------------
__global__ __launch_bounds__(256) void k_conv3(const bf16_t* Esum,
                                               const bf16_t* Woutb,
                                               const float* bout, float* out) {
  int tid = threadIdx.x, w = tid >> 5, lane = tid & 31;
  int n0 = blockIdx.x * 32, b = blockIdx.z;
  int ncol = lane & 15, half = lane >> 4, kh = half;
  int p0 = n0 + ncol, p1 = p0 + 16;
  int y0 = p0 / HD, x0 = p0 % HD;
  int y1 = p1 / HD, x1 = p1 % HD;
  const bf16_t* A0 = Woutb + (size_t)(w * 64) * KTAP;       // wave owns 64 M-rows
  const bf16_t* A1 = A0 + (size_t)16 * KTAP;
  const bf16_t* A2 = A0 + (size_t)32 * KTAP;
  const bf16_t* A3 = A0 + (size_t)48 * KTAP;
  v8f c00 = vzero(), c01 = vzero(), c10 = vzero(), c11 = vzero();
  v8f c20 = vzero(), c21 = vzero(), c30 = vzero(), c31 = vzero();
  for (int tap = 0; tap < 9; ++tap) {
    int dy = tap / 3 - 1, dx = tap % 3 - 1;
    int ny0 = y0 + dy, nx0 = x0 + dx;
    int ny1 = y1 + dy, nx1 = x1 + dx;
    bool ok0 = (ny0 >= 0) & (ny0 < HD) & (nx0 >= 0) & (nx0 < HD);
    bool ok1 = (ny1 >= 0) & (ny1 < HD) & (nx1 >= 0) & (nx1 < HD);
    const bf16_t* br0 =
        Esum + ((size_t)(b * HWP + (ok0 ? ny0 * HD + nx0 : 0))) * CDIM + kh * 16;
    const bf16_t* br1 =
        Esum + ((size_t)(b * HWP + (ok1 ? ny1 * HD + nx1 : 0))) * CDIM + kh * 16;
    __builtin_prefetch(br0, 0, 0);
    __builtin_prefetch(br1, 0, 0);
    int kbase = tap * CDIM;
#pragma unroll 2
    for (int ic0 = 0; ic0 < CDIM; ic0 += 32) {
      v16bf a0 = frag_a(A0, KTAP, kbase + ic0, lane);
      v16bf a1 = frag_a(A1, KTAP, kbase + ic0, lane);
      v16bf a2 = frag_a(A2, KTAP, kbase + ic0, lane);
      v16bf a3 = frag_a(A3, KTAP, kbase + ic0, lane);
      union { v16bf v; uint4 q[2]; } b0, b1;
      if (ok0) {
        b0.q[0] = *(const uint4*)(br0 + ic0);
        b0.q[1] = *(const uint4*)(br0 + ic0 + 8);
      } else {
        b0.q[0] = make_uint4(0u, 0u, 0u, 0u);
        b0.q[1] = make_uint4(0u, 0u, 0u, 0u);
      }
      if (ok1) {
        b1.q[0] = *(const uint4*)(br1 + ic0);
        b1.q[1] = *(const uint4*)(br1 + ic0 + 8);
      } else {
        b1.q[0] = make_uint4(0u, 0u, 0u, 0u);
        b1.q[1] = make_uint4(0u, 0u, 0u, 0u);
      }
      c00 = wmma_bf16(a0, b0.v, c00);
      c01 = wmma_bf16(a0, b1.v, c01);
      c10 = wmma_bf16(a1, b0.v, c10);
      c11 = wmma_bf16(a1, b1.v, c11);
      c20 = wmma_bf16(a2, b0.v, c20);
      c21 = wmma_bf16(a2, b1.v, c21);
      c30 = wmma_bf16(a3, b0.v, c30);
      c31 = wmma_bf16(a3, b1.v, c31);
    }
  }
  v8f accs[4][2] = {{c00, c01}, {c10, c11}, {c20, c21}, {c30, c31}};
#pragma unroll
  for (int i = 0; i < 4; ++i) {
#pragma unroll
    for (int j = 0; j < 2; ++j) {
      int pix = n0 + j * 16 + ncol;
#pragma unroll
      for (int r = 0; r < 8; ++r) {
        int oc = w * 64 + i * 16 + half * 8 + r;
        out[((size_t)(b * CDIM + oc)) * HWP + pix] = accs[i][j][r] + bout[oc];
      }
    }
  }
}

// ---------------------------------------------------------------------------
// Host launcher
// ---------------------------------------------------------------------------
extern "C" void kernel_launch(void* const* d_in, const int* in_sizes, int n_in,
                              void* d_out, int out_size, void* d_ws, size_t ws_size,
                              hipStream_t stream) {
  (void)in_sizes; (void)n_in; (void)out_size; (void)ws_size;
  const float* x    = (const float*)d_in[0];
  const float* WA   = (const float*)d_in[1];
  const float* bA   = (const float*)d_in[2];
  const float* WB   = (const float*)d_in[3];
  const float* bB   = (const float*)d_in[4];
  const float* WV   = (const float*)d_in[5];
  const float* bV   = (const float*)d_in[6];
  const float* WAs  = (const float*)d_in[7];
  const float* bAs  = (const float*)d_in[8];
  const float* WBs  = (const float*)d_in[9];
  const float* bBs  = (const float*)d_in[10];
  const float* WVs  = (const float*)d_in[11];
  const float* bVs  = (const float*)d_in[12];
  const float* WE   = (const float*)d_in[13];
  const float* bE   = (const float*)d_in[14];
  const float* bn_g = (const float*)d_in[15];
  const float* bn_b = (const float*)d_in[16];
  const float* bn_m = (const float*)d_in[17];
  const float* bn_v = (const float*)d_in[18];
  const float* Wout = (const float*)d_in[19];
  const float* bout = (const float*)d_in[20];
  float* out = (float*)d_out;

  // workspace carve-up (256B aligned)
  char* base = (char*)d_ws;
  size_t off = 0;
  auto alloc = [&](size_t bytes) -> char* {
    char* p = base + off;
    off = (off + bytes + 255) & ~(size_t)255;
    return p;
  };
  const size_t NPIX = (size_t)BATCH * HWP;
  bf16_t* xT    = (bf16_t*)alloc(NPIX * CDIM * 2);
  bf16_t* Wcat  = (bf16_t*)alloc((size_t)MQ * CDIM * 2);
  float*  bcat  = (float*) alloc((size_t)MQ * 4);
  bf16_t* WEb   = (bf16_t*)alloc((size_t)CDIM * CM * 2);
  bf16_t* Woutb = (bf16_t*)alloc((size_t)CDIM * KTAP * 2);
  float*  scl   = (float*) alloc((size_t)CDIM * 4);
  float*  shf   = (float*) alloc((size_t)CDIM * 4);
  bf16_t* YA    = (bf16_t*)alloc(NPIX * CM * 2);
  bf16_t* YB    = (bf16_t*)alloc(NPIX * CN * 2);
  bf16_t* YV    = (bf16_t*)alloc(NPIX * CN * 2);
  bf16_t* AsT   = (bf16_t*)alloc(NPIX * CM * 2);
  bf16_t* BsT   = (bf16_t*)alloc(NPIX * CM * 2);
  bf16_t* VsT   = (bf16_t*)alloc(NPIX * CM * 2);
  bf16_t* softB = (bf16_t*)alloc(NPIX * CN * 2);
  bf16_t* softVt= (bf16_t*)alloc(NPIX * CN * 2);
  bf16_t* BsM   = (bf16_t*)alloc(NPIX * CM * 2);
  bf16_t* VsM   = (bf16_t*)alloc(NPIX * CM * 2);
  bf16_t* G     = (bf16_t*)alloc((size_t)BATCH * CM * CN * 2);
  bf16_t* MS    = (bf16_t*)alloc((size_t)BATCH * CM * CM * 2);
  bf16_t* Zt    = (bf16_t*)alloc(NPIX * CM * 2);
  bf16_t* Zst   = (bf16_t*)alloc(NPIX * CM * 2);
  bf16_t* Esum  = (bf16_t*)alloc(NPIX * CDIM * 2);

  // --- weight packing / bn folding ---
  k_pack_wcat<<<(MQ * CDIM + 255) / 256, 256, 0, stream>>>(
      WA, WB, WV, WAs, WBs, WVs, bA, bB, bV, bAs, bBs, bVs, Wcat, bcat);
  k_pack_we<<<(CDIM * CM + 255) / 256, 256, 0, stream>>>(WE, WEb);
  k_pack_wout<<<(CDIM * KTAP + 255) / 256, 256, 0, stream>>>(Wout, Woutb);
  k_bnprep<<<2, 256, 0, stream>>>(bn_g, bn_b, bn_m, bn_v, scl, shf);

  // --- x transpose to channel-major bf16 ---
  k_transpose_x<<<dim3(HWP / 32, CDIM / 32, BATCH), dim3(32, 8), 0, stream>>>(x, xT);

  // --- fused six 1x1 convs (2x2 register-blocked) ---
  k_gemm_qkv<<<dim3(HWP / 32, 5, BATCH), 128, 0, stream>>>(
      xT, Wcat, bcat, YA, YB, YV, AsT, BsT, VsT);

  // --- softmaxes ---
  k_softmax_row<<<BATCH * CN, 256, 0, stream>>>(YB, softB);
  k_softmax_cn64<<<(BATCH * HWP + 255) / 256, 256, 0, stream>>>(YV, softVt);
  k_softmax128<<<(BATCH * HWP + 255) / 256, 256, 0, stream>>>(BsT, BsM);
  k_softmax128<<<(BATCH * HWP + 255) / 256, 256, 0, stream>>>(VsT, VsM);

  // --- attention matmuls (branch 2 via Gram factorization) ---
  k_gemm_G <<<dim3(CN / 16, 1, BATCH), 256, 0, stream>>>(YA, softB, G);
  k_gemm_MS<<<dim3(CM / 16, 1, BATCH), 256, 0, stream>>>(VsM, BsM, MS);
  k_gemm_Z <<<dim3(HWP / 16, 1, BATCH), 256, 0, stream>>>(G, softVt, Zt);
  k_gemm_ZS<<<dim3(HWP / 16, 1, BATCH), 256, 0, stream>>>(MS, AsT, Zst);

  // --- fused WE-GEMMs + BN + ReLU + residual ---
  k_gemm_E<<<dim3(HWP / 16, 4, BATCH), 256, 0, stream>>>(
      Zt, Zst, WEb, bE, scl, shf, xT, Esum);

  // --- 3x3 output conv (implicit GEMM, 64Mx32N per wave) ---
  k_conv3<<<dim3(HWP / 32, 1, BATCH), 256, 0, stream>>>(Esum, Woutb, bout, out);
}
